// MultilayersGAT_35218731828010
// MI455X (gfx1250) — matrix operand order repaired and checked by reading, not compiled, not run on hardware
//
#include <hip/hip_runtime.h>
#include <hip/hip_bf16.h>

// ---------------------------------------------------------------------------
// Multilayer GAT (3 layers) for gfx1250, flash-attention style, bf16 WMMA.
// N=4096 nodes, Fin=512 (all layers), H=4 heads, Fo=128.
//  - Both GEMMs (X@W and att@h) run on v_wmma_f32_16x16x32_bf16, f32 acc.
//  - Online softmax fused into the aggregation (no [H,N,N] materialization).
//  - Software-pipelined loads; conditional (rare) softmax rescale.
//  - Mid-layer elu + head-concat fused into the attention epilogue.
// ---------------------------------------------------------------------------

typedef __attribute__((ext_vector_type(16))) __bf16 v16bf;
typedef __attribute__((ext_vector_type(8)))  __bf16 v8bf;
typedef __attribute__((ext_vector_type(8)))  float  v8f;
typedef __attribute__((ext_vector_type(8)))  int    vi8;

#define NODES 4096
#define FIN   512
#define FO    128
#define HEADS 4
#define NEGBIG (-9e15f)

static __device__ __forceinline__ v16bf cat8(v8bf lo, v8bf hi) {
  v16bf r;
#pragma unroll
  for (int i = 0; i < 8; ++i) { r[i] = lo[i]; r[8 + i] = hi[i]; }
  return r;
}

// ---- fp32 -> bf16 feature convert -----------------------------------------
__global__ void k_convert_x(const float* __restrict__ x, __bf16* __restrict__ xb) {
  int idx = blockIdx.x * blockDim.x + threadIdx.x;   // NODES*FIN
  xb[idx] = (__bf16)x[idx];
}

// ---- W [H][FIN][FO] f32  ->  Wt [H][FO][FIN] bf16 --------------------------
__global__ void k_conv_w(const float* __restrict__ W, __bf16* __restrict__ Wt) {
  int idx  = blockIdx.x * blockDim.x + threadIdx.x;  // HEADS*FIN*FO
  int head = idx / (FIN * FO);
  int rem  = idx - head * (FIN * FO);
  int f    = rem / FO;
  int o    = rem - f * FO;
  Wt[(head * FO + o) * FIN + f] = (__bf16)W[idx];
}

// ---- hT[h][o][n] = sum_f Wt[h][o][f] * X[n][f]   (WMMA bf16, f32 acc) -----
// A = Wt tile (M=fo, K=f, contiguous in f), B = X^T (K=f, N=node).
// Double-buffered over the K loop.
__global__ void k_gemm_xw(const __bf16* __restrict__ X,   // [NODES][FIN]
                          const __bf16* __restrict__ Wt,  // [H][FO][FIN]
                          __bf16* __restrict__ hT) {      // [H][FO][NODES]
  int wave = (blockIdx.x * blockDim.x + threadIdx.x) >> 5;  // 0..2047
  int lane = threadIdx.x & 31;
  int ng   = wave & 63;          // 64-node group
  int ft   = (wave >> 6) & 7;    // 16-fo tile
  int head = wave >> 9;          // 0..3
  int r16  = lane & 15;
  int half = lane >> 4;
  int n0   = ng * 64;
  int kg   = half * 16;

  const __bf16* wrow  = Wt + (size_t)((head * FO) + ft * 16 + r16) * FIN;
  const __bf16* xrow0 = X + (size_t)(n0 + r16) * FIN + half * 16;

  v8f acc[4] = {};

  // prologue: fragments for k0 = 0
  v16bf a_cur = cat8(*(const v8bf*)(wrow + kg), *(const v8bf*)(wrow + kg + 8));
  v16bf b_cur[4];
#pragma unroll
  for (int t = 0; t < 4; ++t)
    b_cur[t] = *(const v16bf*)(xrow0 + (size_t)(t * 16) * FIN);

  for (int k0 = 32; k0 < FIN; k0 += 32) {
    // issue next-iteration loads before this iteration's WMMAs
    v16bf a_nxt = cat8(*(const v8bf*)(wrow + k0 + kg),
                       *(const v8bf*)(wrow + k0 + kg + 8));
    v16bf b_nxt[4];
#pragma unroll
    for (int t = 0; t < 4; ++t)
      b_nxt[t] = *(const v16bf*)(xrow0 + (size_t)(t * 16) * FIN + k0);

#pragma unroll
    for (int t = 0; t < 4; ++t)
      acc[t] = __builtin_amdgcn_wmma_f32_16x16x32_bf16(
          false, a_cur, false, b_cur[t], (short)0, acc[t], false, false);

    a_cur = a_nxt;
#pragma unroll
    for (int t = 0; t < 4; ++t) b_cur[t] = b_nxt[t];
  }
  // epilogue: last K tile
#pragma unroll
  for (int t = 0; t < 4; ++t)
    acc[t] = __builtin_amdgcn_wmma_f32_16x16x32_bf16(
        false, a_cur, false, b_cur[t], (short)0, acc[t], false, false);

#pragma unroll
  for (int t = 0; t < 4; ++t) {
#pragma unroll
    for (int r = 0; r < 8; ++r) {
      int fo   = ft * 16 + r + 8 * half;
      int node = n0 + t * 16 + r16;
      hT[(size_t)(head * FO + fo) * NODES + node] = (__bf16)acc[t][r];
    }
  }
}

// ---- src/dst attention projections ----------------------------------------
__global__ void k_srcdst(const __bf16* __restrict__ hT,  // [H][FO][NODES]
                         const float* __restrict__ a,    // [H][2*FO]
                         float* __restrict__ srcv, float* __restrict__ dstv) {
  int idx  = blockIdx.x * blockDim.x + threadIdx.x;  // HEADS*NODES
  int head = idx >> 12;
  int n    = idx & (NODES - 1);
  const __bf16* base = hT + (size_t)head * FO * NODES + n;
  const float*  ah   = a + head * 2 * FO;
  float s = 0.f, d = 0.f;
#pragma unroll 4
  for (int o = 0; o < FO; ++o) {
    float v = (float)base[(size_t)o * NODES];
    s += v * ah[o];
    d += v * ah[FO + o];
  }
  srcv[idx] = s;
  dstv[idx] = d;
}

// ---- fused masked softmax + (att @ h) via WMMA, online softmax ------------
// One wave per (head, 16-row block). Streams 32-column tiles over all 4096
// neighbors. B fragments load before the softmax VALU block; adj/dst loads
// double-buffered; accumulator rescale only when the running max moved
// (wave-uniform branch, rare after the first few tiles).
// Epilogue: mid layers -> elu + head-concat bf16 (next X);
//           last layer -> normalized f32 hp for the cross-head mean.
__global__ void k_attn(const int* __restrict__ adj,      // [NODES][NODES]
                       const float* __restrict__ srcv,   // [H][NODES]
                       const float* __restrict__ dstv,   // [H][NODES]
                       const __bf16* __restrict__ hT,    // [H][FO][NODES]
                       float* __restrict__ hp,           // [H][NODES][FO]
                       __bf16* __restrict__ Xn,          // [NODES][H*FO]
                       int islast) {
  int head = blockIdx.x >> 8;          // 256 row blocks per head
  int i0   = (blockIdx.x & 255) * 16;
  int lane = threadIdx.x;
  int r16  = lane & 15;
  int half = lane >> 4;
  int cb   = half * 8;

  float src_i = srcv[head * NODES + i0 + r16];
  const int*   arow  = adj + (size_t)(i0 + r16) * NODES;
  const float* drow  = dstv + head * NODES;
  const __bf16* hrow = hT + (size_t)(head * FO + r16) * NODES + half * 8;

  float m = NEGBIG, l = 0.f;
  v8f acc[8] = {};

  // prologue: adj/dst tile for j0 = 0
  vi8 a0 = *(const vi8*)(arow + cb);
  vi8 a1 = *(const vi8*)(arow + cb + 16);
  v8f d0 = *(const v8f*)(drow + cb);
  v8f d1 = *(const v8f*)(drow + cb + 16);

  for (int j0 = 0; j0 < NODES; j0 += 32) {
    // B fragments for this tile, issued early so the softmax VALU work
    // below hides their latency
    v16bf bfrag[8];
#pragma unroll
    for (int c = 0; c < 8; ++c)
      bfrag[c] = cat8(*(const v8bf*)(hrow + (size_t)(c * 16) * NODES + j0),
                      *(const v8bf*)(hrow + (size_t)(c * 16) * NODES + j0 + 16));

    // next-iteration adj/dst (wrap-indexed: stays in-bounds; wrapped data
    // is loaded but never consumed on the final iteration)
    int jn = (j0 + 32) & (NODES - 1);
    vi8 a0n = *(const vi8*)(arow + jn + cb);
    vi8 a1n = *(const vi8*)(arow + jn + cb + 16);
    v8f d0n = *(const v8f*)(drow + jn + cb);
    v8f d1n = *(const v8f*)(drow + jn + cb + 16);

    float e0[8], e1[8];
    float tmax = NEGBIG;
#pragma unroll
    for (int k = 0; k < 8; ++k) {
      float e = src_i + d0[k]; e = (e >= 0.f) ? e : 0.2f * e;
      e0[k] = a0[k] ? e : NEGBIG; tmax = fmaxf(tmax, e0[k]);
      e = src_i + d1[k]; e = (e >= 0.f) ? e : 0.2f * e;
      e1[k] = a1[k] ? e : NEGBIG; tmax = fmaxf(tmax, e1[k]);
    }
    tmax = fmaxf(tmax, __shfl_xor(tmax, 16, 32));   // combine row halves

    // rescale only if any row's running max moved (rare in steady state)
    if (__any(tmax > m)) {
      float mnew  = fmaxf(m, tmax);
      float scale = __expf(m - mnew);
      m = mnew;
      l *= scale;
#pragma unroll
      for (int r = 0; r < 8; ++r) {
        float s = __shfl(scale, r + 8 * half, 32);
#pragma unroll
        for (int c = 0; c < 8; ++c) acc[c][r] *= s;
      }
    }

    float tsum = 0.f;
    v16bf pfrag;
#pragma unroll
    for (int k = 0; k < 8; ++k) {
      float p0 = a0[k] ? __expf(e0[k] - m) : 0.f;
      float p1 = a1[k] ? __expf(e1[k] - m) : 0.f;
      tsum += p0 + p1;
      pfrag[k]     = (__bf16)p0;   // K = cb+k      (A-frag low group)
      pfrag[8 + k] = (__bf16)p1;   // K = 16+cb+k   (A-frag high group)
    }
    tsum += __shfl_xor(tsum, 16, 32);
    l += tsum;

#pragma unroll
    for (int c = 0; c < 8; ++c)
      acc[c] = __builtin_amdgcn_wmma_f32_16x16x32_bf16(
          false, pfrag, false, bfrag[c], (short)0, acc[c], false, false);

    a0 = a0n; a1 = a1n; d0 = d0n; d1 = d1n;
  }

  float linv_lane = 1.0f / l;  // row r16 (dup across halves)
  if (islast) {
    // f32 hp for the cross-head mean
#pragma unroll
    for (int r = 0; r < 8; ++r) {
      float linv = __shfl(linv_lane, r + 8 * half, 32);
      int row = i0 + r + 8 * half;
      float* orow = hp + ((size_t)head * NODES + row) * FO;
#pragma unroll
      for (int c = 0; c < 8; ++c) orow[c * 16 + r16] = acc[c][r] * linv;
    }
  } else {
    // fused elu + head-concat: next layer's bf16 X
#pragma unroll
    for (int r = 0; r < 8; ++r) {
      float linv = __shfl(linv_lane, r + 8 * half, 32);
      int row = i0 + r + 8 * half;
      __bf16* orow = Xn + (size_t)row * (HEADS * FO) + head * FO;
#pragma unroll
      for (int c = 0; c < 8; ++c) {
        float v = acc[c][r] * linv;
        v = (v > 0.f) ? v : (__expf(v) - 1.f);
        orow[c * 16 + r16] = (__bf16)v;
      }
    }
  }
}

// ---- last-layer epilogue: mean over heads + elu -> f32 out -----------------
__global__ void k_epi_last(const float* __restrict__ hp, float* __restrict__ out) {
  int idx = blockIdx.x * blockDim.x + threadIdx.x;   // NODES*FO
  const int HS = NODES * FO;
  float v = 0.25f * (hp[idx] + hp[idx + HS] + hp[idx + 2 * HS] + hp[idx + 3 * HS]);
  out[idx] = (v > 0.f) ? v : (__expf(v) - 1.f);
}

// ---------------------------------------------------------------------------
extern "C" void kernel_launch(void* const* d_in, const int* in_sizes, int n_in,
                              void* d_out, int out_size, void* d_ws, size_t ws_size,
                              hipStream_t stream) {
  const float* x   = (const float*)d_in[0];
  const int*   adj = (const int*)d_in[1];
  const float* W[3] = {(const float*)d_in[2], (const float*)d_in[4], (const float*)d_in[6]};
  const float* A[3] = {(const float*)d_in[3], (const float*)d_in[5], (const float*)d_in[7]};
  float* out = (float*)d_out;

  char* ws = (char*)d_ws;
  __bf16* Xbf  = (__bf16*)(ws);                               //  4 MB
  __bf16* hT   = (__bf16*)(ws + (4u << 20));                  //  4 MB
  __bf16* Wt   = (__bf16*)(ws + (8u << 20));                  // 512 KB
  float*  srcv = (float*) (ws + (8u << 20) + (512u << 10));   //  64 KB
  float*  dstv = (float*) (ws + (8u << 20) + (576u << 10));   //  64 KB
  float*  hp   = (float*) (ws + (9u << 20));                  //  8 MB

  k_convert_x<<<(NODES * FIN) / 256, 256, 0, stream>>>(x, Xbf);

  for (int L = 0; L < 3; ++L) {
    k_conv_w<<<(HEADS * FIN * FO) / 256, 256, 0, stream>>>(W[L], Wt);
    k_gemm_xw<<<512, 128, 0, stream>>>(Xbf, Wt, hT);
    k_srcdst<<<(HEADS * NODES) / 64, 64, 0, stream>>>(hT, A[L], srcv, dstv);
    k_attn<<<HEADS * (NODES / 16), 32, 0, stream>>>(adj, srcv, dstv, hT, hp,
                                                    Xbf, (L == 2) ? 1 : 0);
    if (L == 2)
      k_epi_last<<<(NODES * FO) / 256, 256, 0, stream>>>(hp, out);
  }
}
